// ResidualCorrectionWMv2_21380347199587
// MI455X (gfx1250) — matrix operand order repaired and checked by reading, not compiled, not run on hardware
//
#include <hip/hip_runtime.h>
#include <stdint.h>
#include <stddef.h>

// ---------------- problem constants (from reference) ----------------
#define R_C    3.9f
#define EPS_C  0.3f
#define BETA_C 0.15f
#define NSTEPS 15
#define NB   16
#define CIN  3
#define COUT 3
#define HID  32
#define HH   256
#define WW   256

#define PADW 258          // H+2 / W+2 zero border for 3x3 conv + TDM
#define NCH  32           // nca channels padded 18 -> 32 (K multiple of 32)

// packed weight fragment block: 18 conv1 frags + 2 conv2 frags + 1 conv3 frag
#define W1F_HALFS (18 * 32 * 16)    // 9216
#define W2F_BASE  W1F_HALFS         // 9216
#define W2F_HALFS (2 * 32 * 16)     // 1024
#define W3F_BASE  (W2F_BASE + W2F_HALFS)  // 10240
#define W3F_HALFS (32 * 16)         // 512
#define WF_HALFS  (W3F_BASE + W3F_HALFS)  // 10752

typedef __attribute__((ext_vector_type(16))) _Float16     v16h;
typedef __attribute__((ext_vector_type(8)))  _Float16     v8h;
typedef __attribute__((ext_vector_type(8)))  float        v8f;
typedef __attribute__((ext_vector_type(4)))  unsigned int u32x4;
typedef __attribute__((ext_vector_type(8)))  int          i32x8;
typedef __attribute__((ext_vector_type(4)))  int          i32x4;

union HV { v16h v; v8h h[2]; };

// ---------------- workspace layout ----------------
static constexpr size_t NCA_HALFS  = (size_t)NB * PADW * PADW * NCH;      // 34,080,768
static constexpr size_t NCA_BYTES  = NCA_HALFS * 2;                       // 68,161,536
static constexpr size_t LAST_OFF   = NCA_BYTES;
static constexpr size_t LAST_BYTES = (size_t)NB * COUT * HH * WW * 4;     // 12,582,912
static constexpr size_t WF_OFF     = LAST_OFF + LAST_BYTES;

// TDM only on device pass and only if the builtin exists (6-arg clang-23 form)
#if defined(__HIP_DEVICE_COMPILE__) && defined(__has_builtin)
#if __has_builtin(__builtin_amdgcn_tensor_load_to_lds) && __has_builtin(__builtin_amdgcn_s_wait_tensorcnt)
#define USE_TDM 1
#endif
#endif

__device__ inline v8f wmma_f16(v16h a, v16h b, v8f c) {
  // (neg_a, A, neg_b, B, c_mod, C, reuse_a, reuse_b)
  return __builtin_amdgcn_wmma_f32_16x16x32_f16(false, a, false, b, (short)0, c, false, false);
}

// load one fragment stored fragment-major: [lane][16 halfs] (32B contiguous per lane)
__device__ inline v16h load_frag16(const _Float16* p, int lane) {
  HV u;
  const v8h* q = (const v8h*)(p + lane * 16);
  u.h[0] = q[0];
  u.h[1] = q[1];
  return u.v;
}

// ---------------- kernel 0: zero the nca buffer (border + pad channels) ----
__global__ void zero_nca_kernel(uint4* __restrict__ p, size_t n16) {
  size_t i = (size_t)blockIdx.x * blockDim.x + threadIdx.x;
  size_t stride = (size_t)gridDim.x * blockDim.x;
  uint4 z; z.x = 0; z.y = 0; z.z = 0; z.w = 0;
  for (; i < n16; i += stride) p[i] = z;
}

// ---------------- kernel 1: pack weights into per-lane WMMA B-fragment layout
// Fragment f, lane L, element j  <->  B[K = kbase(L)+j][N = (L&15)], K stored
// contiguous per lane half (lanes 0-15: K 0..15, lanes 16-31: K 16..31).
__global__ void pack_weights_kernel(const float* __restrict__ w1,
                                    const float* __restrict__ w2,
                                    const float* __restrict__ w3,
                                    _Float16* __restrict__ wf) {
  int t = threadIdx.x;
  // conv1: frag = tap*2 + nh (nh: out-channel half)
  for (int idx = t; idx < W1F_HALFS; idx += 256) {
    int frag = idx >> 9;            // /512
    int lane = (idx >> 4) & 31;
    int j    = idx & 15;
    int tap = frag >> 1, nh = frag & 1;
    int ky = tap / 3, kx = tap % 3;
    int kbase = (lane < 16) ? 0 : 16;
    int o = nh * 16 + (lane & 15);
    int i = kbase + j;              // input channel (18 real, pad to 32)
    float v = (i < 18) ? w1[((o * 18 + i) * 3 + ky) * 3 + kx] : 0.f;
    wf[idx] = (_Float16)v;
  }
  // conv2: 2 fragments
  for (int idx = t; idx < W2F_HALFS; idx += 256) {
    int frag = idx >> 9;
    int lane = (idx >> 4) & 31;
    int j    = idx & 15;
    int kbase = (lane < 16) ? 0 : 16;
    int o = frag * 16 + (lane & 15);
    wf[W2F_BASE + idx] = (_Float16)w2[o * 32 + (kbase + j)];
  }
  // conv3: 1 fragment (N padded to 16, only o<3 real)
  for (int idx = t; idx < W3F_HALFS; idx += 256) {
    int lane = (idx >> 4) & 31;
    int j    = idx & 15;
    int kbase = (lane < 16) ? 0 : 16;
    int o = lane & 15;
    wf[W3F_BASE + idx] = (_Float16)((o < 3) ? w3[o * 32 + (kbase + j)] : 0.f);
  }
}

// ---------------- kernel 2: fused 15-step physics in LDS tiles -------------
// One block = one (b,c) 32x32 interior tile; 62x62 region with halo 15.
__global__ __launch_bounds__(256) void physics_kernel(
    const float* __restrict__ x, const float* __restrict__ Kl,
    _Float16* __restrict__ nca, float* __restrict__ lastbuf) {
  __shared__ float gA[62][64];   // current grid
  __shared__ float gB[62][64];   // mapped scratch
  __shared__ float dr[62][64];   // drive (== x)
  const int tid = threadIdx.x;
  const int b = blockIdx.z / COUT, c = blockIdx.z % COUT;
  const int x0 = blockIdx.x * 32, y0 = blockIdx.y * 32;

  const float* kc = Kl + c * 9;
  float k0 = kc[0], k1 = kc[1], k2 = kc[2], k3 = kc[3], k4 = kc[4],
        k5 = kc[5], k6 = kc[6], k7 = kc[7], k8 = kc[8];

  const float* xp = x + ((size_t)(b * CIN + c) * HH) * WW;
  for (int idx = tid; idx < 62 * 62; idx += 256) {
    int i = idx / 62, j = idx % 62;
    int gy = y0 - 15 + i, gx = x0 - 15 + j;
    float v = 0.f;
    if (gy >= 0 && gy < HH && gx >= 0 && gx < WW) v = xp[(size_t)gy * WW + gx];
    gA[i][j] = v; dr[i][j] = v;
  }
  __syncthreads();

  float s[4] = {0.f, 0.f, 0.f, 0.f}, s2[4] = {0.f, 0.f, 0.f, 0.f};
  for (int t = 0; t < NSTEPS; ++t) {
    // phase 1: mapped = R*g*(1-g) on valid region [t, 61-t]
    int lo = t, n = 62 - 2 * t;
    for (int idx = tid; idx < n * n; idx += 256) {
      int i = lo + idx / n, j = lo + idx % n;
      float g = gA[i][j];
      gB[i][j] = R_C * g * (1.f - g);
    }
    __syncthreads();
    // phase 2: physics + clip on [t+1, 60-t]
    int lo2 = t + 1, n2 = n - 2;
    for (int idx = tid; idx < n2 * n2; idx += 256) {
      int i = lo2 + idx / n2, j = lo2 + idx % n2;
      float conv = k0 * gB[i-1][j-1] + k1 * gB[i-1][j] + k2 * gB[i-1][j+1]
                 + k3 * gB[i  ][j-1] + k4 * gB[i  ][j] + k5 * gB[i  ][j+1]
                 + k6 * gB[i+1][j-1] + k7 * gB[i+1][j] + k8 * gB[i+1][j+1];
      float ph = (1.f - EPS_C) * gB[i][j] + EPS_C * conv;
      float g = (1.f - BETA_C) * ph + BETA_C * dr[i][j];
      g = fminf(fmaxf(g, 0.0001f), 1.f - 0.0001f);
      int gy = y0 - 15 + i, gx = x0 - 15 + j;
      if (gy < 0 || gy >= HH || gx < 0 || gx >= WW) g = 0.f;  // zero-pad region
      gA[i][j] = g;
    }
    __syncthreads();
    // accumulate interior stats (cells [15..46]^2)
    #pragma unroll
    for (int k = 0; k < 4; ++k) {
      int idx = tid + k * 256;
      int i = idx >> 5, j = idx & 31;
      float g = gA[15 + i][15 + j];
      s[k] += g; s2[k] += g * g;
    }
    __syncthreads();
  }

  // outputs
  #pragma unroll
  for (int k = 0; k < 4; ++k) {
    int idx = tid + k * 256;
    int i = idx >> 5, j = idx & 31;
    int gy = y0 + i, gx = x0 + j;
    float last  = gA[15 + i][15 + j];
    float drive = dr[15 + i][15 + j];
    float mean  = s[k] * (1.f / NSTEPS);
    float var   = s2[k] * (1.f / NSTEPS) - mean * mean;
    float delta = last - drive;
    lastbuf[((size_t)(b * COUT + c) * HH + gy) * WW + gx] = last;
    size_t base = (((size_t)b * PADW + (gy + 1)) * PADW + (gx + 1)) * NCH;
    nca[base + c]      = (_Float16)drive;   // x channel (drive == x[:, :3])
    nca[base + 3 + c]  = (_Float16)last;
    nca[base + 6 + c]  = (_Float16)mean;
    nca[base + 9 + c]  = (_Float16)var;
    nca[base + 12 + c] = (_Float16)delta;   // delta
    nca[base + 15 + c] = (_Float16)delta;   // last_drive == delta
  }
}

// ---------------- kernel 3: fused conv1(WMMA)+relu+conv2(WMMA)+relu+conv3(WMMA)
// Block = 128 pixels of one image row; 8 waves x (16 px x 32 ch).
__global__ __launch_bounds__(256) void fused_nn_kernel(
    const _Float16* __restrict__ nca, const float* __restrict__ lastbuf,
    const _Float16* __restrict__ wfg, const float* __restrict__ b1,
    const float* __restrict__ b2, const float* __restrict__ b3,
    float* __restrict__ out) {
  struct Smem {
    alignas(16) _Float16 nca[3 * 130 * NCH];  // MUST stay first: TDM lds_addr == 0
    alignas(16) _Float16 wf[WF_HALFS];        // fragment-major packed weights
    alignas(16) _Float16 h[8][16 * 32];       // per-wave activation re-layout scratch
  };
  __shared__ Smem sh;
  const int tid  = threadIdx.x;
  const int lane = tid & 31;
  const int wv   = tid >> 5;
  const int x0   = blockIdx.x * 128;
  const int y    = blockIdx.y;
  const int b    = blockIdx.z;

  // stage nca tile: padded-buffer rows y..y+2, cols x0..x0+129 (image y-1..y+1, x0-1..x0+128)
  const size_t tile_off = (((size_t)b * PADW + y) * PADW + x0) * NCH;  // halfs
#ifdef USE_TDM
  if (wv == 0) {
    unsigned long long ga = (unsigned long long)(uintptr_t)nca + tile_off * 2ull;
    u32x4 g0;
    g0[0] = 1u;                                        // count=1, user load
    g0[1] = 0u;                                        // lds_addr = 0 (sh.nca)
    g0[2] = (unsigned)(ga & 0xFFFFFFFFull);
    g0[3] = (unsigned)((ga >> 32) & 0x1FFFFFFull) | (2u << 30);  // type=2
    i32x8 g1;
    g1[0] = 0x00010000;                                // data_size=2B, no mask/pad
    g1[1] = (int)(((PADW * NCH) & 0xFFFF) << 16);      // tensor_dim0 lo16
    g1[2] = (int)(PADW << 16);                         // tensor_dim1 lo16
    g1[3] = (int)((130 * NCH) << 16);                  // tile_dim0 = 4160 elems
    g1[4] = 3;                                         // tile_dim1 = 3 rows
    g1[5] = PADW * NCH;                                // tensor_dim0_stride
    g1[6] = 0; g1[7] = 0;
    i32x4 gz4; gz4[0] = 0; gz4[1] = 0; gz4[2] = 0; gz4[3] = 0;   // groups 2/3 unused (2D)
    i32x8 gz8; gz8[0] = 0; gz8[1] = 0; gz8[2] = 0; gz8[3] = 0;
    gz8[4] = 0; gz8[5] = 0; gz8[6] = 0; gz8[7] = 0;
    __builtin_amdgcn_tensor_load_to_lds(g0, g1, gz4, gz4, gz8, 0);
    __builtin_amdgcn_s_wait_tensorcnt(0);
  }
#else
  {
    const unsigned* src = (const unsigned*)(nca + tile_off);
    unsigned* dst = (unsigned*)sh.nca;
    const int rowDW = (PADW * NCH) / 2;                // 4128 dwords/global row
    for (int k = tid; k < 3 * 130 * NCH / 2; k += 256) {
      int row = k / (130 * NCH / 2);
      int off = k % (130 * NCH / 2);
      dst[k] = src[(size_t)row * rowDW + off];
    }
  }
#endif
  // weights: contiguous 21.5 KB block -> LDS via uint4
  {
    const uint4* src = (const uint4*)wfg;
    uint4* dst = (uint4*)sh.wf;
    for (int k = tid; k < WF_HALFS / 8; k += 256) dst[k] = src[k];
  }
  __syncthreads();

  // WMMA fragment index helpers (wave32 16-bit; K contiguous per lane half)
  const int m     = lane & 15;                 // A: M row, B/C: N col
  const int khalf = (lane < 16) ? 0 : 16;      // A fragment K split
  const int px    = wv * 16 + m;               // pixel within 128-block

  v8f acc0 = {0.f,0.f,0.f,0.f,0.f,0.f,0.f,0.f};
  v8f acc1 = {0.f,0.f,0.f,0.f,0.f,0.f,0.f,0.f};

  // conv1: implicit GEMM, K = 9 taps x 32 padded channels
  #pragma unroll
  for (int tap = 0; tap < 9; ++tap) {
    const int dy = tap / 3, dx = tap % 3;
    const _Float16* asrc = &sh.nca[(dy * 130 + (px + dx)) * NCH];
    HV ua;
    ua.h[0] = *(const v8h*)(asrc + khalf);        // K 0..7 / 16..23
    ua.h[1] = *(const v8h*)(asrc + 8 + khalf);    // K 8..15 / 24..31
    v16h bA = load_frag16(&sh.wf[(tap * 2 + 0) * 512], lane);
    v16h bB = load_frag16(&sh.wf[(tap * 2 + 1) * 512], lane);
    acc0 = wmma_f16(ua.v, bA, acc0);
    acc1 = wmma_f16(ua.v, bB, acc1);
  }

  // bias + relu, re-layout C->A through per-wave LDS scratch
  _Float16* hbuf = sh.h[wv];
  {
    float bb0 = b1[m], bb1 = b1[16 + m];
    #pragma unroll
    for (int r = 0; r < 8; ++r) {
      int mm = r + ((lane < 16) ? 0 : 8);
      float v0 = acc0[r] + bb0; v0 = v0 > 0.f ? v0 : 0.f;
      float v1 = acc1[r] + bb1; v1 = v1 > 0.f ? v1 : 0.f;
      hbuf[mm * 32 + m]      = (_Float16)v0;
      hbuf[mm * 32 + 16 + m] = (_Float16)v1;
    }
  }
  HV u2;
  u2.h[0] = *(const v8h*)(hbuf + m * 32 + khalf);
  u2.h[1] = *(const v8h*)(hbuf + m * 32 + 8 + khalf);

  // conv2: 1x1 32->32, single K=32 step
  v16h b2A = load_frag16(&sh.wf[W2F_BASE], lane);
  v16h b2B = load_frag16(&sh.wf[W2F_BASE + 512], lane);
  v8f zf = {0.f,0.f,0.f,0.f,0.f,0.f,0.f,0.f};
  v8f c20 = wmma_f16(u2.v, b2A, zf);
  v8f c21 = wmma_f16(u2.v, b2B, zf);
  {
    float bb0 = b2[m], bb1 = b2[16 + m];
    #pragma unroll
    for (int r = 0; r < 8; ++r) {
      int mm = r + ((lane < 16) ? 0 : 8);
      float v0 = c20[r] + bb0; v0 = v0 > 0.f ? v0 : 0.f;
      float v1 = c21[r] + bb1; v1 = v1 > 0.f ? v1 : 0.f;
      hbuf[mm * 32 + m]      = (_Float16)v0;
      hbuf[mm * 32 + 16 + m] = (_Float16)v1;
    }
  }
  HV u3;
  u3.h[0] = *(const v8h*)(hbuf + m * 32 + khalf);
  u3.h[1] = *(const v8h*)(hbuf + m * 32 + 8 + khalf);

  // conv3: 1x1 32->3 (N padded to 16)
  v16h b3f = load_frag16(&sh.wf[W3F_BASE], lane);
  v8f c3 = wmma_f16(u3.v, b3f, zf);

  // out = clip(last + correction, 0, 1); only N<3 lanes write
  #pragma unroll
  for (int r = 0; r < 8; ++r) {
    int mm = r + ((lane < 16) ? 0 : 8);
    if (m < 3) {
      int gx = x0 + wv * 16 + mm;
      size_t li = (((size_t)b * COUT + m) * HH + y) * WW + gx;
      float v = lastbuf[li] + (c3[r] + b3[m]);
      v = fminf(fmaxf(v, 0.f), 1.f);
      out[li] = v;
    }
  }
}

// ---------------- host launcher ----------------
extern "C" void kernel_launch(void* const* d_in, const int* in_sizes, int n_in,
                              void* d_out, int out_size, void* d_ws, size_t ws_size,
                              hipStream_t stream) {
  (void)in_sizes; (void)n_in; (void)out_size; (void)ws_size;
  const float* x  = (const float*)d_in[0];
  const float* Kl = (const float*)d_in[1];
  const float* w1 = (const float*)d_in[2];
  const float* b1 = (const float*)d_in[3];
  const float* w2 = (const float*)d_in[4];
  const float* b2 = (const float*)d_in[5];
  const float* w3 = (const float*)d_in[6];
  const float* b3 = (const float*)d_in[7];
  float* out = (float*)d_out;

  char* ws = (char*)d_ws;
  _Float16* nca     = (_Float16*)ws;
  float*    lastbuf = (float*)(ws + LAST_OFF);
  _Float16* wf      = (_Float16*)(ws + WF_OFF);

  zero_nca_kernel<<<2048, 256, 0, stream>>>((uint4*)nca, NCA_BYTES / 16);
  pack_weights_kernel<<<1, 256, 0, stream>>>(w1, w2, w3, wf);
  physics_kernel<<<dim3(WW / 32, HH / 32, NB * COUT), 256, 0, stream>>>(x, Kl, nca, lastbuf);
  fused_nn_kernel<<<dim3(WW / 128, HH, NB), 256, 0, stream>>>(
      nca, lastbuf, wf, b1, b2, b3, out);
}
